// _SAGE_52432960750005
// MI455X (gfx1250) — matrix operand order repaired and checked by reading, not compiled, not run on hardware
//
#include <hip/hip_runtime.h>
#include <hip/hip_bf16.h>
#include <stdint.h>

#define N_NODES 100000
#define N_EDGES 1600000
#define NCFG    8
#define HID     64
#define OPED    64
#define NFD     140
#define CFD     18
#define NGRAPH  16
#define NROW    (N_NODES * NCFG)   /* 800000 rows of (node,config) */
#define SLOTS   (NCFG * HID)       /* 512 values per node */
#define LALPHA  0.2f
#define KPAD0   160                /* 140 padded to multiple of 32 */
#define PCHUNK  32                 /* pooling chunks per graph */

typedef __attribute__((ext_vector_type(16))) __bf16       v16bf;
typedef __attribute__((ext_vector_type(8)))  float        v8f;
typedef __attribute__((ext_vector_type(4)))  unsigned int u32x4;
typedef __attribute__((ext_vector_type(2)))  unsigned int u32x2;

union V16U {
  unsigned short s[16];
  u32x4          q[2];
  v16bf          v;
};

__device__ __forceinline__ unsigned short f2bf(float f) {
  unsigned u = __float_as_uint(f);
  unsigned r = u + 0x7fffu + ((u >> 16) & 1u);   // round-to-nearest-even
  return (unsigned short)(r >> 16);
}
__device__ __forceinline__ float bf2f(unsigned short h) {
  return __uint_as_float(((unsigned)h) << 16);
}
__device__ __forceinline__ float leakyf(float x) { return x > 0.f ? x : LALPHA * x; }

// ---------------- utility fills / converts ----------------

__global__ void k_fill_u32(unsigned* p, unsigned v, int n) {
  int i = blockIdx.x * blockDim.x + threadIdx.x;
  if (i < n) p[i] = v;
}

// node_feats [N,140] f32 -> [N,160] bf16 (zero padded)
__global__ void k_cvt_pad_nf(const float* __restrict__ nf, unsigned short* __restrict__ out) {
  int i = blockIdx.x * blockDim.x + threadIdx.x;
  if (i >= N_NODES * KPAD0) return;
  int n = i / KPAD0, k = i - n * KPAD0;
  float v = (k < NFD) ? nf[(size_t)n * NFD + k] : 0.f;
  out[i] = f2bf(v);
}

// opd[o,h] = sum_k op_emb[o,k] * W_d0[k,h]  (rows 0..63 of W), same for p
__global__ void k_opproj(const float* __restrict__ op_emb,
                         const float* __restrict__ Wd0, const float* __restrict__ Wp0,
                         float* __restrict__ opd, float* __restrict__ opp) {
  int i = blockIdx.x * blockDim.x + threadIdx.x;
  if (i >= 120 * HID) return;
  int o = i >> 6, h = i & 63;
  float sd = 0.f, sp = 0.f;
  for (int k = 0; k < OPED; k++) {
    float e = op_emb[o * OPED + k];
    sd += e * Wd0[k * HID + h];
    sp += e * Wp0[k * HID + h];
  }
  opd[i] = sd;
  opp[i] = sp;
}

// ---------------- WMMA helpers (CDNA5 v_wmma_f32_16x16x32_bf16) ----------------

// A 16x32 bf16 tile: lane<16 -> row, K chunks {0..7} and {16..23}; lane>=16 -> {8..15},{24..31}
__device__ __forceinline__ v16bf load_a_tile(const unsigned short* __restrict__ X, int ld,
                                             int rowBase, int kBase, int lane) {
  int r  = rowBase + (lane & 15);
  int k0 = kBase + ((lane >> 4) & 1) * 8;
  const unsigned short* pb = X + (size_t)r * ld + k0;
  V16U u;
  u.q[0] = *(const u32x4*)(pb);
  u.q[1] = *(const u32x4*)(pb + 16);
  return u.v;
}

// B 32x16 bf16 tile from f32 weight W[k,h] (row-major, ld=64): lane gives col; K = hi*8 + j (j<8), 16 + hi*8 + (j-8) (j>=8)
__device__ __forceinline__ v16bf build_b_tile(const float* __restrict__ W, int kBase, int kMax,
                                              int colBase, int lane) {
  V16U u;
  int col = colBase + (lane & 15);
  int kb  = kBase + ((lane >> 4) & 1) * 8;
#pragma unroll
  for (int j = 0; j < 16; j++) {
    int k = kb + (j < 8 ? j : j + 8);
    float w = (k < kMax) ? W[k * HID + col] : 0.f;
    u.s[j] = f2bf(w);
  }
  return u.v;
}

// nodeproj = nf_bf[N,160] @ W_mid[140(pad160),64], two weight matrices at once (f32 out)
__global__ void k_node_gemm(const unsigned short* __restrict__ Abf,
                            const float* __restrict__ WdMid, const float* __restrict__ WpMid,
                            float* __restrict__ outd, float* __restrict__ outp) {
  int lane   = threadIdx.x & 31;
  int wave   = blockIdx.x * (blockDim.x >> 5) + (threadIdx.x >> 5);
  int nwaves = gridDim.x * (blockDim.x >> 5);
  const int NT = N_NODES / 16;   // 6250 row tiles
  int hi = (lane >> 4) & 1;
  for (int ct = 0; ct < 4; ct++) {
    v16bf Bd[5], Bp[5];
#pragma unroll
    for (int ks = 0; ks < 5; ks++) {
      Bd[ks] = build_b_tile(WdMid, ks * 32, NFD, ct * 16, lane);
      Bp[ks] = build_b_tile(WpMid, ks * 32, NFD, ct * 16, lane);
    }
    int col = ct * 16 + (lane & 15);
    for (int tile = wave; tile < NT; tile += nwaves) {
      int rowBase = tile * 16;
      // load all A k-step tiles up front so loads overlap the WMMA chain
      v16bf a[5];
#pragma unroll
      for (int ks = 0; ks < 5; ks++) a[ks] = load_a_tile(Abf, KPAD0, rowBase, ks * 32, lane);
      v8f accd = {0, 0, 0, 0, 0, 0, 0, 0};
      v8f accp = {0, 0, 0, 0, 0, 0, 0, 0};
#pragma unroll
      for (int ks = 0; ks < 5; ks++)
        accd = __builtin_amdgcn_wmma_f32_16x16x32_bf16(false, a[ks], false, Bd[ks], (short)0, accd, false, false);
#pragma unroll
      for (int ks = 0; ks < 5; ks++)
        accp = __builtin_amdgcn_wmma_f32_16x16x32_bf16(false, a[ks], false, Bp[ks], (short)0, accp, false, false);
      // one base address per output tile; constant 256B row stride folds into IOFFSET
      float* pd = outd + (size_t)(rowBase + 8 * hi) * HID + col;
      float* pp = outp + (size_t)(rowBase + 8 * hi) * HID + col;
#pragma unroll
      for (int r = 0; r < 8; r++) pd[r * HID] = accd[r];
#pragma unroll
      for (int r = 0; r < 8; r++) pp[r * HID] = accp[r];
    }
  }
}

// per-(node,config) K=18 config projection + combine with node/op parts
// t (bf16) = pre-SpMM projected features (x @ Wd0); p (f32) = x @ Wp0 + bp0
__global__ void k_cfgproj(const float* __restrict__ cfg, const int* __restrict__ op_ids,
                          const float* __restrict__ opd, const float* __restrict__ opp,
                          const float* __restrict__ ndp_d, const float* __restrict__ ndp_p,
                          const float* __restrict__ Wd0, const float* __restrict__ Wp0,
                          const float* __restrict__ bp0,
                          unsigned short* __restrict__ t, float* __restrict__ p) {
  int gi  = blockIdx.x * blockDim.x + threadIdx.x;   // over NROW*64
  int row = gi >> 6;                                  // (n*8+c)
  int h   = gi & 63;
  if (row >= NROW) return;
  int n = row >> 3;
  const float* cf = cfg + (size_t)row * CFD;
  int opid = op_ids[n];
  const float* WdB = Wd0 + (size_t)(OPED + NFD) * HID;  // rows 204..221
  const float* WpB = Wp0 + (size_t)(OPED + NFD) * HID;
  float sd = ndp_d[(size_t)n * HID + h] + opd[opid * HID + h];
  float sp = ndp_p[(size_t)n * HID + h] + opp[opid * HID + h] + bp0[h];
#pragma unroll
  for (int k = 0; k < CFD; k++) {
    float c = cf[k];
    sd += c * WdB[k * HID + h];
    sp += c * WpB[k * HID + h];
  }
  t[(size_t)row * HID + h] = f2bf(sd);
  p[(size_t)row * HID + h] = sp;
}

// ---------------- CSR build (A + A^T), deterministic after sort ----------------

__global__ void k_deg(const int* __restrict__ src, const int* __restrict__ dst, int* __restrict__ deg) {
  int e = blockIdx.x * blockDim.x + threadIdx.x;
  if (e >= N_EDGES) return;
  atomicAdd(&deg[dst[e]], 1);
  atomicAdd(&deg[src[e]], 1);
}

__global__ void k_scan(const int* __restrict__ deg, int* __restrict__ rowptr, int n) {
  __shared__ int buf[2][1024];
  int carry = 0;
  if (threadIdx.x == 0) rowptr[0] = 0;
  for (int base = 0; base < n; base += 1024) {
    int i = base + (int)threadIdx.x;
    int v = (i < n) ? deg[i] : 0;
    int a = 0;
    buf[0][threadIdx.x] = v;
    __syncthreads();
    for (int off = 1; off < 1024; off <<= 1) {
      int tv = buf[a][threadIdx.x];
      if ((int)threadIdx.x >= off) tv += buf[a][threadIdx.x - off];
      __syncthreads();
      buf[1 - a][threadIdx.x] = tv;
      a = 1 - a;
      __syncthreads();
    }
    if (i < n) rowptr[i + 1] = carry + buf[a][threadIdx.x];
    carry += buf[a][1023];
    __syncthreads();
  }
}

__global__ void k_csrfill(const int* __restrict__ src, const int* __restrict__ dst,
                          const int* __restrict__ rowptr, int* __restrict__ cursor,
                          int* __restrict__ col) {
  int e = blockIdx.x * blockDim.x + threadIdx.x;
  if (e >= N_EDGES) return;
  int s = src[e], d = dst[e];
  int p1 = atomicAdd(&cursor[d], 1);
  col[rowptr[d] + p1] = s;
  int p2 = atomicAdd(&cursor[s], 1);
  col[rowptr[s] + p2] = d;
}

// deterministic order: insertion-sort each adjacency list (avg degree 32)
__global__ void k_sortadj(const int* __restrict__ rowptr, int* __restrict__ col) {
  int n = blockIdx.x * blockDim.x + threadIdx.x;
  if (n >= N_NODES) return;
  int b = rowptr[n], e = rowptr[n + 1];
  for (int i = b + 1; i < e; i++) {
    int v = col[i];
    int j = i - 1;
    while (j >= b && col[j] > v) { col[j + 1] = col[j]; j--; }
    col[j + 1] = v;
  }
}

// ---------------- layer GEMM (layers 1,2): x[800000,64] bf16 -> t bf16, p f32 ----------------

__global__ void k_layer_gemm(const unsigned short* __restrict__ x,
                             const float* __restrict__ Wd, const float* __restrict__ Wp,
                             const float* __restrict__ bp,
                             unsigned short* __restrict__ t, float* __restrict__ p) {
  int lane   = threadIdx.x & 31;
  int wave   = blockIdx.x * (blockDim.x >> 5) + (threadIdx.x >> 5);
  int nwaves = gridDim.x * (blockDim.x >> 5);
  int hi = (lane >> 4) & 1;
  int lcol = lane & 15;
  v16bf B[2][4][2];
#pragma unroll
  for (int m = 0; m < 2; m++)
#pragma unroll
    for (int ct = 0; ct < 4; ct++)
#pragma unroll
      for (int ks = 0; ks < 2; ks++)
        B[m][ct][ks] = build_b_tile(m ? Wp : Wd, ks * 32, HID, ct * 16, lane);
  float bpc[4];
#pragma unroll
  for (int ct = 0; ct < 4; ct++) bpc[ct] = bp[ct * 16 + lcol];

  const int NT = NROW / 16;   // 50000 row tiles
  for (int tile = wave; tile < NT; tile += nwaves) {
    int rowBase = tile * 16;
    v16bf a0 = load_a_tile(x, HID, rowBase, 0, lane);
    v16bf a1 = load_a_tile(x, HID, rowBase, 32, lane);
    size_t obase = (size_t)(rowBase + 8 * hi) * HID + lcol;
#pragma unroll
    for (int m = 0; m < 2; m++) {
#pragma unroll
      for (int ct = 0; ct < 4; ct++) {
        v8f acc = {0, 0, 0, 0, 0, 0, 0, 0};
        acc = __builtin_amdgcn_wmma_f32_16x16x32_bf16(false, a0, false, B[m][ct][0], (short)0, acc, false, false);
        acc = __builtin_amdgcn_wmma_f32_16x16x32_bf16(false, a1, false, B[m][ct][1], (short)0, acc, false, false);
        if (m == 0) {
          unsigned short* pt = t + obase + ct * 16;
#pragma unroll
          for (int r = 0; r < 8; r++) pt[r * HID] = f2bf(acc[r]);
        } else {
          float* pp = p + obase + ct * 16;
          float bb = bpc[ct];
#pragma unroll
          for (int r = 0; r < 8; r++) pp[r * HID] = acc[r] + bb;
        }
      }
    }
  }
}

// ---------------- fused SpMM + epilogue: y=adj(t); x = l2norm(leaky(y+bd) + p) ----------------

__global__ void k_spmm_layer(const int* __restrict__ rowptr, const int* __restrict__ col,
                             const unsigned short* __restrict__ t, const float* __restrict__ p,
                             const float* __restrict__ bd, unsigned short* __restrict__ xout) {
  int n   = blockIdx.x;        // one node per block
  int tid = threadIdx.x;       // 128 threads: 4 slots each (512 total)
  int b = rowptr[n], e = rowptr[n + 1];
  float a0 = 0.f, a1 = 0.f, a2 = 0.f, a3 = 0.f;
  float c0 = 0.f, c1 = 0.f, c2 = 0.f, c3 = 0.f;
  int i = b;
  int soff = tid * 4;
  // unroll x2 with independent accumulator sets: fixed, deterministic reduction tree
  for (; i + 1 < e; i += 2) {
    int j0 = col[i], j1 = col[i + 1];
    int jp = col[(i + 2 < e) ? (i + 2) : (i + 1)];
    __builtin_prefetch(t + (size_t)jp * SLOTS + soff, 0, 0);     // global_prefetch_b8
    u32x2 d0 = *(const u32x2*)(t + (size_t)j0 * SLOTS + soff);
    u32x2 d1 = *(const u32x2*)(t + (size_t)j1 * SLOTS + soff);
    a0 += bf2f((unsigned short)(d0.x & 0xffffu));
    a1 += bf2f((unsigned short)(d0.x >> 16));
    a2 += bf2f((unsigned short)(d0.y & 0xffffu));
    a3 += bf2f((unsigned short)(d0.y >> 16));
    c0 += bf2f((unsigned short)(d1.x & 0xffffu));
    c1 += bf2f((unsigned short)(d1.x >> 16));
    c2 += bf2f((unsigned short)(d1.y & 0xffffu));
    c3 += bf2f((unsigned short)(d1.y >> 16));
  }
  if (i < e) {
    int j0 = col[i];
    u32x2 d0 = *(const u32x2*)(t + (size_t)j0 * SLOTS + soff);
    a0 += bf2f((unsigned short)(d0.x & 0xffffu));
    a1 += bf2f((unsigned short)(d0.x >> 16));
    a2 += bf2f((unsigned short)(d0.y & 0xffffu));
    a3 += bf2f((unsigned short)(d0.y >> 16));
  }
  a0 += c0; a1 += c1; a2 += c2; a3 += c3;
  int f0 = soff & 63;
  size_t o = (size_t)n * SLOTS + soff;
  float v0 = leakyf(a0 + bd[f0 + 0]) + p[o + 0];
  float v1 = leakyf(a1 + bd[f0 + 1]) + p[o + 1];
  float v2 = leakyf(a2 + bd[f0 + 2]) + p[o + 2];
  float v3 = leakyf(a3 + bd[f0 + 3]) + p[o + 3];
  // l2norm over each 64-wide config group = 16 consecutive lanes within the wave
  float sq = v0 * v0 + v1 * v1 + v2 * v2 + v3 * v3;
  sq += __shfl_xor(sq, 1);
  sq += __shfl_xor(sq, 2);
  sq += __shfl_xor(sq, 4);
  sq += __shfl_xor(sq, 8);
  float rs = rsqrtf(fmaxf(sq, 1e-12f));
  u32x2 st;
  st.x = (unsigned)f2bf(v0 * rs) | ((unsigned)f2bf(v1 * rs) << 16);
  st.y = (unsigned)f2bf(v2 * rs) | ((unsigned)f2bf(v3 * rs) << 16);
  *(u32x2*)(xout + o) = st;    // single 8B store
}

// ---------------- pooling (graph_ids sorted -> atomics-free, deterministic) ----------------

__global__ void k_graph_bounds(const int* __restrict__ gids, int* __restrict__ gstart) {
  int g = threadIdx.x;
  if (g > NGRAPH) return;
  int lo = 0, hi = N_NODES;
  while (lo < hi) {
    int mid = (lo + hi) >> 1;
    if (gids[mid] < g) lo = mid + 1; else hi = mid;
  }
  gstart[g] = lo;
}

// stage 1: fixed chunking of each graph's node range -> partial sums
__global__ void k_pool_part(const unsigned short* __restrict__ x, const int* __restrict__ gstart,
                            float* __restrict__ part) {
  int g  = blockIdx.x / PCHUNK;
  int ch = blockIdx.x % PCHUNK;
  int slot = threadIdx.x;      // 512
  int b = gstart[g], e = gstart[g + 1];
  int len = e - b;
  int per = (len + PCHUNK - 1) / PCHUNK;
  int s0 = b + ch * per;
  int s1 = s0 + per; if (s1 > e) s1 = e;
  float acc = 0.f;
  for (int n = s0; n < s1; n++) acc += bf2f(x[(size_t)n * SLOTS + slot]);
  part[((size_t)g * PCHUNK + ch) * SLOTS + slot] = acc;
}

// stage 2: in-order reduction of the chunks (deterministic)
__global__ void k_pool_reduce(const float* __restrict__ part, float* __restrict__ pooled) {
  int g = blockIdx.x, slot = threadIdx.x;
  float acc = 0.f;
  for (int ch = 0; ch < PCHUNK; ch++)
    acc += part[((size_t)g * PCHUNK + ch) * SLOTS + slot];
  pooled[g * SLOTS + slot] = acc;
}

// ---------------- tiny postnet MLP: [128,64] -> 64 -> 64 -> 1 ----------------

__global__ void k_mlp(const float* __restrict__ pooled,
                      const float* __restrict__ Wm0, const float* __restrict__ bm0,
                      const float* __restrict__ Wm1, const float* __restrict__ bm1,
                      const float* __restrict__ Wm2, const float* __restrict__ bm2,
                      float* __restrict__ out) {
  __shared__ float v[128][64];
  __shared__ float h1[128][65];
  int row = threadIdx.x;    // 128 rows = B*C
  for (int k = 0; k < 64; k++) v[row][k] = pooled[row * 64 + k];
  for (int h = 0; h < 64; h++) {
    float s = bm0[h];
    for (int k = 0; k < 64; k++) s += v[row][k] * Wm0[k * 64 + h];
    h1[row][h] = leakyf(s);
  }
  float o = bm2[0];
  for (int h = 0; h < 64; h++) {
    float s = bm1[h];
    for (int k = 0; k < 64; k++) s += h1[row][k] * Wm1[k * 64 + h];
    o += leakyf(s) * Wm2[h];
  }
  out[row] = o;
}

// ---------------- host orchestration ----------------

extern "C" void kernel_launch(void* const* d_in, const int* in_sizes, int n_in,
                              void* d_out, int out_size, void* d_ws, size_t ws_size,
                              hipStream_t stream) {
  (void)in_sizes; (void)n_in; (void)out_size; (void)ws_size;
  const float* node_feats   = (const float*)d_in[0];
  const float* config_feats = (const float*)d_in[1];
  const int*   op_ids       = (const int*)d_in[2];
  const int*   src          = (const int*)d_in[3];
  const int*   dst          = (const int*)d_in[4];
  const int*   gids         = (const int*)d_in[5];
  const float* op_emb       = (const float*)d_in[6];
  const float* W_d0 = (const float*)d_in[7];  const float* b_d0 = (const float*)d_in[8];
  const float* W_p0 = (const float*)d_in[9];  const float* b_p0 = (const float*)d_in[10];
  const float* W_d1 = (const float*)d_in[11]; const float* b_d1 = (const float*)d_in[12];
  const float* W_p1 = (const float*)d_in[13]; const float* b_p1 = (const float*)d_in[14];
  const float* W_d2 = (const float*)d_in[15]; const float* b_d2 = (const float*)d_in[16];
  const float* W_p2 = (const float*)d_in[17]; const float* b_p2 = (const float*)d_in[18];
  const float* W_m0 = (const float*)d_in[19]; const float* b_m0 = (const float*)d_in[20];
  const float* W_m1 = (const float*)d_in[21]; const float* b_m1 = (const float*)d_in[22];
  const float* W_m2 = (const float*)d_in[23]; const float* b_m2 = (const float*)d_in[24];
  float* out = (float*)d_out;

  char* wsp = (char*)d_ws;
  auto alloc = [&](size_t bytes) -> char* {
    char* r = wsp;
    wsp += (bytes + 255) & ~(size_t)255;
    return r;
  };
  unsigned short* nf_bf  = (unsigned short*)alloc((size_t)N_NODES * KPAD0 * 2);
  float*          opd    = (float*)alloc((size_t)120 * HID * 4);
  float*          opp    = (float*)alloc((size_t)120 * HID * 4);
  float*          ndp_d  = (float*)alloc((size_t)N_NODES * HID * 4);
  float*          ndp_p  = (float*)alloc((size_t)N_NODES * HID * 4);
  unsigned short* t      = (unsigned short*)alloc((size_t)NROW * HID * 2);
  float*          p      = (float*)alloc((size_t)NROW * HID * 4);
  unsigned short* x      = (unsigned short*)alloc((size_t)NROW * HID * 2);
  int*            deg    = (int*)alloc((size_t)N_NODES * 4);
  int*            cursor = (int*)alloc((size_t)N_NODES * 4);
  int*            rowptr = (int*)alloc((size_t)(N_NODES + 1) * 4);
  int*            colidx = (int*)alloc((size_t)2 * N_EDGES * 4);
  float*          part   = (float*)alloc((size_t)NGRAPH * PCHUNK * SLOTS * 4);
  float*          pooled = (float*)alloc((size_t)NGRAPH * SLOTS * 4);
  int*            gstart = (int*)alloc((size_t)(NGRAPH + 1) * 4);

  // zero the counters used by atomics
  k_fill_u32<<<(N_NODES + 255) / 256, 256, 0, stream>>>((unsigned*)deg, 0u, N_NODES);
  k_fill_u32<<<(N_NODES + 255) / 256, 256, 0, stream>>>((unsigned*)cursor, 0u, N_NODES);

  // layer-0 projections (decomposed: op table / node GEMM / config K=18)
  k_cvt_pad_nf<<<(N_NODES * KPAD0 + 255) / 256, 256, 0, stream>>>(node_feats, nf_bf);
  k_opproj<<<(120 * HID + 255) / 256, 256, 0, stream>>>(op_emb, W_d0, W_p0, opd, opp);
  k_node_gemm<<<200, 256, 0, stream>>>(nf_bf, W_d0 + OPED * HID, W_p0 + OPED * HID, ndp_d, ndp_p);
  k_cfgproj<<<(NROW * 64) / 256, 256, 0, stream>>>(config_feats, op_ids, opd, opp, ndp_d, ndp_p,
                                                   W_d0, W_p0, b_p0, t, p);

  // CSR of A + A^T, deterministic after per-row sort
  k_deg<<<(N_EDGES + 255) / 256, 256, 0, stream>>>(src, dst, deg);
  k_scan<<<1, 1024, 0, stream>>>(deg, rowptr, N_NODES);
  k_csrfill<<<(N_EDGES + 255) / 256, 256, 0, stream>>>(src, dst, rowptr, cursor, colidx);
  k_sortadj<<<(N_NODES + 255) / 256, 256, 0, stream>>>(rowptr, colidx);

  // layer 0
  k_spmm_layer<<<N_NODES, 128, 0, stream>>>(rowptr, colidx, t, p, b_d0, x);
  // layer 1
  k_layer_gemm<<<512, 256, 0, stream>>>(x, W_d1, W_p1, b_p1, t, p);
  k_spmm_layer<<<N_NODES, 128, 0, stream>>>(rowptr, colidx, t, p, b_d1, x);
  // layer 2
  k_layer_gemm<<<512, 256, 0, stream>>>(x, W_d2, W_p2, b_p2, t, p);
  k_spmm_layer<<<N_NODES, 128, 0, stream>>>(rowptr, colidx, t, p, b_d2, x);

  // pool + MLP head (two-stage, atomics-free, deterministic)
  k_graph_bounds<<<1, 32, 0, stream>>>(gids, gstart);
  k_pool_part<<<NGRAPH * PCHUNK, SLOTS, 0, stream>>>(x, gstart, part);
  k_pool_reduce<<<NGRAPH, SLOTS, 0, stream>>>(part, pooled);
  k_mlp<<<1, 128, 0, stream>>>(pooled, W_m0, b_m0, W_m1, b_m1, W_m2, b_m2, out);
}
